// DeformableTransformer_14602888806449
// MI455X (gfx1250) — matrix operand order repaired and checked by reading, not compiled, not run on hardware
//
#include <hip/hip_runtime.h>
#include <hip/hip_bf16.h>

// ---------------------------------------------------------------------------
// Types for CDNA5 WMMA
// ---------------------------------------------------------------------------
typedef __attribute__((ext_vector_type(16))) __bf16 v16bf;
typedef __attribute__((ext_vector_type(8)))  float  v8f;

union FragU {
    v16bf v;
    uint4 u[2];
};

// f32 -> bf16 (round-to-nearest-even), raw bits (avoids relying on __bf16 arith)
static __device__ __forceinline__ unsigned short f2bf(float f) {
    unsigned u = __float_as_uint(f);
    u += 0x7FFFu + ((u >> 16) & 1u);
    return (unsigned short)(u >> 16);
}

// ---------------------------------------------------------------------------
// Problem constants (from the reference)
// ---------------------------------------------------------------------------
#define NTOK 52500      // B * L = 4 * 13125
#define LTOT 13125
#define DMODEL 256

// GEMM tiling
#define BM 128
#define BN 64
#define BK 32
#define LSTR 40         // LDS row stride in bf16 elems (32 data + 8 pad), 80B, 16B aligned

// ---------------------------------------------------------------------------
// Tiled WMMA GEMM:  Out[M,Ncols] = act(A[M,K] (+A2) @ W[K,Ncols] + bias)
//   ABF16  : A is raw bf16 (unsigned short), else f32 (converted on load)
//   ADDPOS : A := A + A2 elementwise on load (used for q = src + pos)
//   RELU   : ReLU epilogue
//   OUTBF16: store raw bf16 instead of f32
// ---------------------------------------------------------------------------
template <bool ABF16, bool ADDPOS, bool RELU, bool OUTBF16>
__global__ __launch_bounds__(256) void gemm_wmma(
    const void* __restrict__ Av, const float* __restrict__ A2,
    const float* __restrict__ W, const float* __restrict__ bias,
    void* __restrict__ Out, int M, int K, int Ncols)
{
    __shared__ unsigned short As[BM * LSTR];  // 10 KB
    __shared__ unsigned short Bs[BN * LSTR];  //  5 KB

    const int tid  = threadIdx.x;
    const int lane = tid & 31;
    const int wave = tid >> 5;
    const int wm   = wave >> 1;   // 0..3  -> 32-row sub-tile
    const int wn   = wave & 1;    // 0..1  -> 32-col sub-tile
    const int m0   = blockIdx.x * BM;
    const int n0   = blockIdx.y * BN;

    const float*          Af = (const float*)Av;
    const unsigned short* Ab = (const unsigned short*)Av;

    v8f acc[2][2];
#pragma unroll
    for (int tm = 0; tm < 2; ++tm)
#pragma unroll
        for (int tn = 0; tn < 2; ++tn)
#pragma unroll
            for (int e = 0; e < 8; ++e) acc[tm][tn][e] = 0.0f;

    for (int kk = 0; kk < K; kk += BK) {
        // ---- stage A tile [BM x BK] into LDS (bf16) ----
        if (ABF16) {
#pragma unroll
            for (int it = 0; it < 2; ++it) {
                int e = it * 256 + tid;        // 0..511
                int r = e >> 2;                // 0..127
                int c = (e & 3) * 8;           // 0,8,16,24
                uint4 v = {0u, 0u, 0u, 0u};
                if (m0 + r < M)
                    v = *(const uint4*)(Ab + (size_t)(m0 + r) * K + kk + c);
                *(uint4*)(&As[r * LSTR + c]) = v;
            }
        } else {
#pragma unroll
            for (int it = 0; it < 4; ++it) {
                int e = it * 256 + tid;        // 0..1023
                int r = e >> 3;                // 0..127
                int c = (e & 7) * 4;           // 0..28
                float4 f = {0.f, 0.f, 0.f, 0.f};
                if (m0 + r < M) {
                    const float* ap = Af + (size_t)(m0 + r) * K + kk + c;
                    f = *(const float4*)ap;
                    if (ADDPOS) {
                        float4 p = *(const float4*)(A2 + (size_t)(m0 + r) * K + kk + c);
                        f.x += p.x; f.y += p.y; f.z += p.z; f.w += p.w;
                    }
                    if (kk + BK < K) __builtin_prefetch(ap + BK, 0, 3);
                }
                unsigned short* d = &As[r * LSTR + c];
                d[0] = f2bf(f.x); d[1] = f2bf(f.y); d[2] = f2bf(f.z); d[3] = f2bf(f.w);
            }
        }
        // ---- stage B tile transposed: Bs[n][k], n in [0,BN), k in [0,BK) ----
#pragma unroll
        for (int it = 0; it < 2; ++it) {
            int e  = it * 256 + tid;           // 0..511
            int kr = e >> 4;                   // 0..31
            int c4 = (e & 15) * 4;             // 0..60
            const float* wp = W + (size_t)(kk + kr) * Ncols + n0 + c4;
#pragma unroll
            for (int j = 0; j < 4; ++j) {
                float f = (n0 + c4 + j < Ncols) ? wp[j] : 0.0f;
                Bs[(c4 + j) * LSTR + kr] = f2bf(f);
            }
        }
        __syncthreads();

        // ---- load fragments per CDNA5 16-bit A/B lane layouts ----
        FragU a[2], b[2];
        const int koff = (lane >> 4) * 8;      // A: K half-groups of 8
        const int kb0  = (lane >> 4) * 8;      // B: K half-groups of 8
#pragma unroll
        for (int tm = 0; tm < 2; ++tm) {
            int r = wm * 32 + tm * 16 + (lane & 15);
            a[tm].u[0] = *(const uint4*)(&As[r * LSTR + koff]);
            a[tm].u[1] = *(const uint4*)(&As[r * LSTR + 16 + koff]);
        }
#pragma unroll
        for (int tn = 0; tn < 2; ++tn) {
            int c = wn * 32 + tn * 16 + (lane & 15);
            b[tn].u[0] = *(const uint4*)(&Bs[c * LSTR + kb0]);
            b[tn].u[1] = *(const uint4*)(&Bs[c * LSTR + kb0 + 16]);
        }
#pragma unroll
        for (int tm = 0; tm < 2; ++tm)
#pragma unroll
            for (int tn = 0; tn < 2; ++tn)
                acc[tm][tn] = __builtin_amdgcn_wmma_f32_16x16x32_bf16(
                    false, a[tm].v, false, b[tn].v, (short)0, acc[tm][tn],
                    false, false);
        __syncthreads();
    }

    // ---- epilogue: bias (+ReLU) and store ----
#pragma unroll
    for (int tm = 0; tm < 2; ++tm)
#pragma unroll
        for (int tn = 0; tn < 2; ++tn)
#pragma unroll
            for (int e = 0; e < 8; ++e) {
                int row = m0 + wm * 32 + tm * 16 + (lane >> 4) * 8 + e;
                int col = n0 + wn * 32 + tn * 16 + (lane & 15);
                if (row < M && col < Ncols) {
                    float v = acc[tm][tn][e] + bias[col];
                    if (RELU) v = fmaxf(v, 0.0f);
                    if (OUTBF16)
                        ((unsigned short*)Out)[(size_t)row * Ncols + col] = f2bf(v);
                    else
                        ((float*)Out)[(size_t)row * Ncols + col] = v;
                }
            }
}

// ---------------------------------------------------------------------------
// In-place softmax over 12 (NL*NP) logits per (token, head)
// ---------------------------------------------------------------------------
__global__ __launch_bounds__(256) void softmax12(float* __restrict__ a, int R) {
    int r = blockIdx.x * blockDim.x + threadIdx.x;
    if (r >= R) return;
    float* p = a + (size_t)r * 12;
    float m = p[0];
#pragma unroll
    for (int i = 1; i < 12; ++i) m = fmaxf(m, p[i]);
    float e[12], s = 0.f;
#pragma unroll
    for (int i = 0; i < 12; ++i) { e[i] = __expf(p[i] - m); s += e[i]; }
    float inv = 1.0f / s;
#pragma unroll
    for (int i = 0; i < 12; ++i) p[i] = e[i] * inv;
}

// ---------------------------------------------------------------------------
// Deformable attention sampling. One wave per (token, head); lane = channel.
// value : [B, L, NH, HD] f32        offl : [N, 192] f32 (pre-scale offsets)
// attn  : [N, NH, 12] f32 (softmaxed)  ref : [B, L, NL, 2] f32
// out   : sampled [N, 256] bf16 (head-major channels)
// ---------------------------------------------------------------------------
__global__ __launch_bounds__(256) void deform_sample(
    const float* __restrict__ value, const float* __restrict__ offl,
    const float* __restrict__ attn,  const float* __restrict__ ref,
    unsigned short* __restrict__ sampled)
{
    const int n    = blockIdx.x;        // token (grid = NTOK)
    const int head = threadIdx.x >> 5;  // 8 waves = 8 heads
    const int lane = threadIdx.x & 31;  // HD channel
    const int b    = n / LTOT;

    const int Hs[3] = {100, 50, 25};
    const int Ws[3] = {100, 50, 25};
    const int St[3] = {0, 10000, 12500};

    const float* offp = offl + (size_t)n * 192 + head * 24;           // NL*NP*2
    const float* ap   = attn + ((size_t)n * 8 + head) * 12;

    float acc = 0.0f;
#pragma unroll
    for (int lid = 0; lid < 3; ++lid) {
        const int Hh = Hs[lid], Ww = Ws[lid], st = St[lid];
        const float rx = ref[((size_t)n * 3 + lid) * 2 + 0];
        const float ry = ref[((size_t)n * 3 + lid) * 2 + 1];
#pragma unroll
        for (int p = 0; p < 4; ++p) {
            float ox = offp[(lid * 4 + p) * 2 + 0];
            float oy = offp[(lid * 4 + p) * 2 + 1];
            // loc*dim - 0.5  (align_corners=False), offsets pre-divided by dim
            float x = rx * (float)Ww + ox - 0.5f;
            float y = ry * (float)Hh + oy - 0.5f;
            float x0f = floorf(x), y0f = floorf(y);
            float wx1 = x - x0f, wy1 = y - y0f;
            int ix0 = (int)x0f, iy0 = (int)y0f;
            float aw  = ap[lid * 4 + p];
            float w00 = (1.f - wx1) * (1.f - wy1) * aw;
            float w10 = wx1 * (1.f - wy1) * aw;
            float w01 = (1.f - wx1) * wy1 * aw;
            float w11 = wx1 * wy1 * aw;
#pragma unroll
            for (int t = 0; t < 4; ++t) {
                int ix = ix0 + (t & 1);
                int iy = iy0 + (t >> 1);
                float w = (t == 0) ? w00 : (t == 1) ? w10 : (t == 2) ? w01 : w11;
                if (ix >= 0 && ix < Ww && iy >= 0 && iy < Hh) {
                    size_t vrow = (size_t)b * LTOT + st + (size_t)iy * Ww + ix;
                    acc += w * value[(vrow * 8 + head) * 32 + lane];
                }
            }
        }
    }
    sampled[(size_t)n * 256 + head * 32 + lane] = f2bf(acc);
}

// ---------------------------------------------------------------------------
// out = LayerNorm(in1 + in2) * g + b ; one wave32 per token, shfl reduction
// ---------------------------------------------------------------------------
__global__ __launch_bounds__(256) void resid_ln(
    const float* __restrict__ in1, const float* __restrict__ in2,
    const float* __restrict__ g,   const float* __restrict__ bta,
    float* __restrict__ out, int M)
{
    int wave = (blockIdx.x * blockDim.x + threadIdx.x) >> 5;
    int lane = threadIdx.x & 31;
    if (wave >= M) return;
    const float* p1 = in1 + (size_t)wave * DMODEL;
    const float* p2 = in2 + (size_t)wave * DMODEL;
    float v[8], s1 = 0.f, s2 = 0.f;
#pragma unroll
    for (int j = 0; j < 8; ++j) {
        int i = lane + 32 * j;
        float t = p1[i] + p2[i];
        v[j] = t; s1 += t; s2 += t * t;
    }
#pragma unroll
    for (int m = 16; m; m >>= 1) {
        s1 += __shfl_xor(s1, m, 32);
        s2 += __shfl_xor(s2, m, 32);
    }
    float mean = s1 * (1.0f / DMODEL);
    float var  = s2 * (1.0f / DMODEL) - mean * mean;
    float rstd = rsqrtf(var + 1e-5f);
    float* o = out + (size_t)wave * DMODEL;
#pragma unroll
    for (int j = 0; j < 8; ++j) {
        int i = lane + 32 * j;
        o[i] = (v[j] - mean) * rstd * g[i] + bta[i];
    }
}

// ---------------------------------------------------------------------------
// Host launch
// ---------------------------------------------------------------------------
extern "C" void kernel_launch(void* const* d_in, const int* in_sizes, int n_in,
                              void* d_out, int out_size, void* d_ws, size_t ws_size,
                              hipStream_t stream) {
    const float* src   = (const float*)d_in[0];
    const float* pos   = (const float*)d_in[1];
    const float* refp  = (const float*)d_in[2];
    const float* Wv    = (const float*)d_in[3];
    const float* bv    = (const float*)d_in[4];
    const float* Woff  = (const float*)d_in[5];
    const float* boff  = (const float*)d_in[6];
    const float* Wattn = (const float*)d_in[7];
    const float* battn = (const float*)d_in[8];
    const float* Wout  = (const float*)d_in[9];
    const float* bout  = (const float*)d_in[10];
    const float* g1    = (const float*)d_in[11];
    const float* b1    = (const float*)d_in[12];
    const float* Wff1  = (const float*)d_in[13];
    const float* bff1  = (const float*)d_in[14];
    const float* Wff2  = (const float*)d_in[15];
    const float* bff2  = (const float*)d_in[16];
    const float* g2    = (const float*)d_in[17];
    const float* b2    = (const float*)d_in[18];

    // Workspace layout (phase-based reuse):
    //   [value f32 | offl f32 | attn f32]   <- phase A; phase C reuses as h(bf16)
    //   [sampled bf16][out1/ff f32][x f32]
    char* ws = (char*)d_ws;
    const size_t szValue = (size_t)NTOK * 256 * 4;   // 53.76 MB
    const size_t szOffl  = (size_t)NTOK * 192 * 4;   // 40.32 MB
    const size_t szAttn  = (size_t)NTOK * 96  * 4;   // 20.16 MB
    const size_t szSamp  = (size_t)NTOK * 256 * 2;   // 26.88 MB
    const size_t offOffl = szValue;
    const size_t offAttn = offOffl + szOffl;
    const size_t offSamp = offAttn + szAttn;         // end of region 0 (114.24 MB >= h 107.52 MB)
    const size_t offOut1 = offSamp + szSamp;
    const size_t offX    = offOut1 + szValue;

    float*          valueb = (float*)(ws);
    float*          offlb  = (float*)(ws + offOffl);
    float*          attnb  = (float*)(ws + offAttn);
    unsigned short* sampb  = (unsigned short*)(ws + offSamp);
    float*          out1b  = (float*)(ws + offOut1);
    float*          xb     = (float*)(ws + offX);
    unsigned short* hb     = (unsigned short*)(ws);          // reuses region 0
    float*          ffb    = (float*)(ws + offOut1);         // reuses out1

    const dim3 blk(256);
    const int  mb = (NTOK + BM - 1) / BM;   // 411

    // value = src @ Wv + bv
    gemm_wmma<false, false, false, false><<<dim3(mb, 4), blk, 0, stream>>>(
        src, nullptr, Wv, bv, valueb, NTOK, 256, 256);
    // off logits = (src+pos) @ Woff + boff
    gemm_wmma<false, true, false, false><<<dim3(mb, 3), blk, 0, stream>>>(
        src, pos, Woff, boff, offlb, NTOK, 256, 192);
    // attn logits = (src+pos) @ Wattn + battn
    gemm_wmma<false, true, false, false><<<dim3(mb, 2), blk, 0, stream>>>(
        src, pos, Wattn, battn, attnb, NTOK, 256, 96);
    // softmax over 12 per (token, head)
    softmax12<<<(NTOK * 8 + 255) / 256, blk, 0, stream>>>(attnb, NTOK * 8);
    // deformable bilinear sampling -> bf16 [N,256]
    deform_sample<<<NTOK, blk, 0, stream>>>(valueb, offlb, attnb, refp, sampb);
    // attn_out = sampled @ Wout + bout
    gemm_wmma<true, false, false, false><<<dim3(mb, 4), blk, 0, stream>>>(
        sampb, nullptr, Wout, bout, out1b, NTOK, 256, 256);
    // x = LN(src + attn_out)
    resid_ln<<<(NTOK + 7) / 8, blk, 0, stream>>>(src, out1b, g1, b1, xb, NTOK);
    // h = relu(x @ Wff1 + bff1) -> bf16 [N,1024]
    gemm_wmma<false, false, true, true><<<dim3(mb, 16), blk, 0, stream>>>(
        xb, nullptr, Wff1, bff1, hb, NTOK, 256, 1024);
    // ff = h @ Wff2 + bff2
    gemm_wmma<true, false, false, false><<<dim3(mb, 4), blk, 0, stream>>>(
        hb, nullptr, Wff2, bff2, ffb, NTOK, 1024, 256);
    // out = LN(x + ff)
    resid_ln<<<(NTOK + 7) / 8, blk, 0, stream>>>(xb, ffb, g2, b2, (float*)d_out, NTOK);
}
